// GAT_Net_Class_31172872635027
// MI455X (gfx1250) — matrix operand order repaired and checked by reading, not compiled, not run on hardware
//
#include <hip/hip_runtime.h>
#include <math.h>

static constexpr int N_NODES = 100000;
static constexpr int E_EDGES = 1600000;
static constexpr int E_TOT   = E_EDGES + N_NODES;   // edges + self loops
static constexpr float NEG_SLOPE = 0.2f;

typedef __attribute__((ext_vector_type(16))) _Float16 v16h;
typedef __attribute__((ext_vector_type(8)))  float    v8f;

// ---- order-preserving float <-> uint key (for atomicMax on signed floats) ----
__device__ __forceinline__ unsigned f32_key(float f) {
  unsigned u = __float_as_uint(f);
  return (u & 0x80000000u) ? ~u : (u | 0x80000000u);
}
__device__ __forceinline__ float key_f32(unsigned k) {
  return (k & 0x80000000u) ? __uint_as_float(k ^ 0x80000000u)
                           : __uint_as_float(~k);
}
// key of -inf (0xFF800000) == ~0xFF800000 == 0x007FFFFF
static constexpr unsigned NEG_INF_KEY = 0x007FFFFFu;

// ------------------------------ fills ---------------------------------------
__global__ void fill_u32_kernel(unsigned* __restrict__ p, unsigned v, int n) {
  int i = blockIdx.x * blockDim.x + threadIdx.x;
  if (i < n) p[i] = v;
}

// W [K,C] f32 row-major  ->  Wt [C,K] f16 (each output column contiguous)
__global__ void convert_wt_kernel(const float* __restrict__ W,
                                  _Float16* __restrict__ Wt, int K, int C) {
  int i = blockIdx.x * blockDim.x + threadIdx.x;
  if (i >= K * C) return;
  int n = i / K, k = i - n * K;
  Wt[(size_t)n * K + k] = (_Float16)W[(size_t)k * C + n];
}

// --------------------------- WMMA GEMM --------------------------------------
// C[M,Ncols] = A[M,K](f32) * B[K,Ncols], B given pre-transposed as Bt[Ncols,K] f16.
// grid.x = M/16 (M multiple of 16); block = 32 * (Ncols/16): one wave per 16-col tile.
// K multiple of 32. f16 multiply, f32 accumulate via v_wmma_f32_16x16x32_f16.
__global__ void wmma_gemm_kernel(const float* __restrict__ A,
                                 const _Float16* __restrict__ Bt,
                                 float* __restrict__ C, int K, int Ncols) {
  const int lane = threadIdx.x & 31;
  const int wave = threadIdx.x >> 5;          // column tile index
  const int m    = lane & 15;
  const int hi   = lane >> 4;                 // 0 or 1 (lane half)
  const int row0 = blockIdx.x * 16;

  // ISA 7.12.2 16-bit A 16x32: lane half 'hi' holds K = hi*8..hi*8+7 and 16+hi*8..16+hi*8+7
  const float*    arow = A  + (size_t)(row0 + m) * K + hi * 8;
  // B 32x16: lane half 'hi' holds K = hi*8..hi*8+7 and 16+hi*8..16+hi*8+7 of column N
  const _Float16* brow = Bt + (size_t)(wave * 16 + m) * K + hi * 8;

  v8f acc = {};
  for (int kk = 0; kk < K; kk += 32) {
    const float4* ap0 = (const float4*)(arow + kk);
    const float4* ap1 = (const float4*)(arow + kk + 16);
    float4 f0 = ap0[0], f1 = ap0[1];
    float4 f2 = ap1[0], f3 = ap1[1];
    v16h a;
    a[0]=(_Float16)f0.x;  a[1]=(_Float16)f0.y;  a[2]=(_Float16)f0.z;  a[3]=(_Float16)f0.w;
    a[4]=(_Float16)f1.x;  a[5]=(_Float16)f1.y;  a[6]=(_Float16)f1.z;  a[7]=(_Float16)f1.w;
    a[8]=(_Float16)f2.x;  a[9]=(_Float16)f2.y;  a[10]=(_Float16)f2.z; a[11]=(_Float16)f2.w;
    a[12]=(_Float16)f3.x; a[13]=(_Float16)f3.y; a[14]=(_Float16)f3.z; a[15]=(_Float16)f3.w;
    typedef __attribute__((ext_vector_type(8))) _Float16 v8h_;
    v8h_ b0 = *(const v8h_*)(brow + kk), b1 = *(const v8h_*)(brow + kk + 16);
    v16h b = __builtin_shufflevector(b0, b1, 0,1,2,3,4,5,6,7,8,9,10,11,12,13,14,15);
    acc = __builtin_amdgcn_wmma_f32_16x16x32_f16(false, a, false, b,
                                                 (short)0, acc, false, false);
  }
  // D layout: lane half hi -> rows M = hi*8 + r, col N = wave*16 + (lane&15)
  float* crow = C + (size_t)(row0 + hi * 8) * Ncols + wave * 16 + m;
#pragma unroll
  for (int r = 0; r < 8; ++r) crow[(size_t)r * Ncols] = acc[r];
}

// --------------------------- attention alphas -------------------------------
// as[n,h] = sum_f h[n,h,f]*a_src[h,f]; ad likewise. One thread per (node, head).
__global__ void alpha_kernel(const float* __restrict__ h,
                             const float* __restrict__ a_src,
                             const float* __restrict__ a_dst,
                             float* __restrict__ as, float* __restrict__ ad,
                             int H, int F) {
  int i = blockIdx.x * blockDim.x + threadIdx.x;
  if (i >= N_NODES * H) return;
  int n = i / H, hh = i - n * H;
  const float* hp  = h + ((size_t)n * H + hh) * F;
  const float* sp  = a_src + hh * F;
  const float* dp  = a_dst + hh * F;
  float ss = 0.f, dd = 0.f;
  for (int f = 0; f < F; ++f) { float v = hp[f]; ss += v * sp[f]; dd += v * dp[f]; }
  as[i] = ss; ad[i] = dd;
}

// ------------------------ edge pass 1: logits + seg-max ----------------------
__global__ void edge_max_kernel(const int* __restrict__ ei,
                                const float* __restrict__ as,
                                const float* __restrict__ ad,
                                float* __restrict__ elog,
                                unsigned* __restrict__ mkey, int H) {
  int i = blockIdx.x * blockDim.x + threadIdx.x;
  if (i >= E_TOT * H) return;
  int e = i / H, hh = i - e * H;
  int s, d;
  if (e < E_EDGES) { s = ei[e]; d = ei[E_EDGES + e]; } else { s = d = e - E_EDGES; }
  float v = as[s * H + hh] + ad[d * H + hh];
  v = v > 0.f ? v : NEG_SLOPE * v;            // leaky_relu
  elog[i] = v;
  atomicMax(mkey + (size_t)d * H + hh, f32_key(v));
}

__global__ void decode_max_kernel(unsigned* __restrict__ k, int n) {
  int i = blockIdx.x * blockDim.x + threadIdx.x;
  if (i < n) { unsigned v = k[i]; ((float*)k)[i] = key_f32(v); }
}

// ------------------------ edge pass 2: exp + seg-sum -------------------------
__global__ void edge_sum_kernel(const int* __restrict__ ei,
                                const float* __restrict__ m,
                                float* __restrict__ elog,
                                float* __restrict__ ssum, int H) {
  int i = blockIdx.x * blockDim.x + threadIdx.x;
  if (i >= E_TOT * H) return;
  int e = i / H, hh = i - e * H;
  int d = (e < E_EDGES) ? ei[E_EDGES + e] : (e - E_EDGES);
  float ex = __expf(elog[i] - m[(size_t)d * H + hh]);
  elog[i] = ex;                                // overwrite logit with exp value
  atomicAdd(ssum + (size_t)d * H + hh, ex);
}

// ------------------- edge pass 3: alpha-weighted scatter ---------------------
__global__ void edge_agg_kernel(const int* __restrict__ ei,
                                const float* __restrict__ elog,
                                const float* __restrict__ denom,
                                const float* __restrict__ h,
                                float* __restrict__ out, int H, int F) {
  int i = blockIdx.x * blockDim.x + threadIdx.x;
  if (i >= E_TOT * H) return;
  int e = i / H, hh = i - e * H;
  int s, d;
  if (e < E_EDGES) { s = ei[e]; d = ei[E_EDGES + e]; } else { s = d = e - E_EDGES; }
  float alpha = elog[i] / denom[(size_t)d * H + hh];
  const float4* hp = (const float4*)(h + ((size_t)s * H + hh) * F);
  float* op = out + ((size_t)d * H + hh) * F;
  for (int q = 0; q < (F >> 2); ++q) {
    float4 v = hp[q];
    atomicAdd(op + 4 * q + 0, alpha * v.x);
    atomicAdd(op + 4 * q + 1, alpha * v.y);
    atomicAdd(op + 4 * q + 2, alpha * v.z);
    atomicAdd(op + 4 * q + 3, alpha * v.w);
  }
}

// --------------------------- bias + ELU (in place) ---------------------------
__global__ void elu_bias_kernel(float* __restrict__ o, const float* __restrict__ b,
                                int C, int total) {
  int i = blockIdx.x * blockDim.x + threadIdx.x;
  if (i >= total) return;
  float v = o[i] + b[i % C];
  o[i] = v > 0.f ? v : (__expf(v) - 1.0f);
}

// --------------------------- final log_softmax -------------------------------
__global__ void logsoftmax_kernel(const float* __restrict__ o,
                                  const float* __restrict__ b,
                                  float* __restrict__ out) {
  int i = blockIdx.x * blockDim.x + threadIdx.x;
  if (i >= N_NODES) return;
  const float* p = o + (size_t)i * 32;
  float vals[32];
  float mx = -INFINITY;
#pragma unroll
  for (int c = 0; c < 32; ++c) { float v = p[c] + b[c]; vals[c] = v; mx = fmaxf(mx, v); }
  float s = 0.f;
#pragma unroll
  for (int c = 0; c < 32; ++c) s += __expf(vals[c] - mx);
  float lg = __logf(s);
  float* op = out + (size_t)i * 32;
#pragma unroll
  for (int c = 0; c < 32; ++c) op[c] = vals[c] - mx - lg;
}

// ============================================================================
extern "C" void kernel_launch(void* const* d_in, const int* in_sizes, int n_in,
                              void* d_out, int out_size, void* d_ws, size_t ws_size,
                              hipStream_t stream) {
  (void)in_sizes; (void)n_in; (void)out_size; (void)ws_size;

  const float* x       = (const float*)d_in[0];   // [N,256]
  const int*   ei      = (const int*)d_in[1];     // [2,E]
  const float* W1      = (const float*)d_in[2];   // [256,128]
  const float* attS1   = (const float*)d_in[3];   // [8,16]
  const float* attD1   = (const float*)d_in[4];   // [8,16]
  const float* b1      = (const float*)d_in[5];   // [128]
  const float* W2      = (const float*)d_in[6];   // [128,32]
  const float* attS2   = (const float*)d_in[7];   // [1,32]
  const float* attD2   = (const float*)d_in[8];   // [1,32]
  const float* b2      = (const float*)d_in[9];   // [32]
  float*       out     = (float*)d_out;           // [N,32]

  // ---------------- workspace carve-out (256B aligned) ----------------
  char* base = (char*)d_ws;
  size_t off = 0;
  auto take = [&](size_t bytes) -> void* {
    void* p = base + off;
    off += (bytes + 255) & ~(size_t)255;
    return p;
  };
  _Float16* w1t  = (_Float16*)take((size_t)128 * 256 * 2);
  _Float16* w2t  = (_Float16*)take((size_t)32 * 128 * 2);
  float*    h1   = (float*)take((size_t)N_NODES * 128 * 4);
  float*    o1   = (float*)take((size_t)N_NODES * 128 * 4);
  float*    as1  = (float*)take((size_t)N_NODES * 8 * 4);
  float*    ad1  = (float*)take((size_t)N_NODES * 8 * 4);
  unsigned* m1   = (unsigned*)take((size_t)N_NODES * 8 * 4);  // keys, then f32
  float*    s1   = (float*)take((size_t)N_NODES * 8 * 4);
  float*    e1   = (float*)take((size_t)E_TOT * 8 * 4);
  float*    h2   = (float*)take((size_t)N_NODES * 32 * 4);
  float*    o2   = (float*)take((size_t)N_NODES * 32 * 4);
  float*    as2  = (float*)take((size_t)N_NODES * 4);
  float*    ad2  = (float*)take((size_t)N_NODES * 4);
  unsigned* m2   = (unsigned*)take((size_t)N_NODES * 4);
  float*    s2   = (float*)take((size_t)N_NODES * 4);
  float*    e2   = (float*)take((size_t)E_TOT * 4);

  const int T = 256;
  auto blk = [](int n, int t) { return (n + t - 1) / t; };

  // ---------------- init accumulators (every call: ws is poisoned) ----
  fill_u32_kernel<<<blk(N_NODES*8,T),T,0,stream>>>(m1, NEG_INF_KEY, N_NODES*8);
  fill_u32_kernel<<<blk(N_NODES*8,T),T,0,stream>>>((unsigned*)s1, 0u, N_NODES*8);
  fill_u32_kernel<<<blk(N_NODES*128,T),T,0,stream>>>((unsigned*)o1, 0u, N_NODES*128);
  fill_u32_kernel<<<blk(N_NODES,T),T,0,stream>>>(m2, NEG_INF_KEY, N_NODES);
  fill_u32_kernel<<<blk(N_NODES,T),T,0,stream>>>((unsigned*)s2, 0u, N_NODES);
  fill_u32_kernel<<<blk(N_NODES*32,T),T,0,stream>>>((unsigned*)o2, 0u, N_NODES*32);

  // ---------------- weight transpose + f16 convert --------------------
  convert_wt_kernel<<<blk(256*128,T),T,0,stream>>>(W1, w1t, 256, 128);
  convert_wt_kernel<<<blk(128*32,T),T,0,stream>>>(W2, w2t, 128, 32);

  // ======================= layer 1 (H=8, F=16) ========================
  // h1 = x @ W1   (WMMA, 8 waves/block -> 16x128 tile per block)
  wmma_gemm_kernel<<<N_NODES/16, 256, 0, stream>>>(x, w1t, h1, 256, 128);
  alpha_kernel<<<blk(N_NODES*8,T),T,0,stream>>>(h1, attS1, attD1, as1, ad1, 8, 16);
  edge_max_kernel<<<blk(E_TOT*8,T),T,0,stream>>>(ei, as1, ad1, e1, m1, 8);
  decode_max_kernel<<<blk(N_NODES*8,T),T,0,stream>>>(m1, N_NODES*8);
  edge_sum_kernel<<<blk(E_TOT*8,T),T,0,stream>>>(ei, (const float*)m1, e1, s1, 8);
  edge_agg_kernel<<<blk(E_TOT*8,T),T,0,stream>>>(ei, e1, s1, h1, o1, 8, 16);
  elu_bias_kernel<<<blk(N_NODES*128,T),T,0,stream>>>(o1, b1, 128, N_NODES*128);

  // ======================= layer 2 (H=1, F=32) ========================
  // h2 = elu_out @ W2   (WMMA, 2 waves/block -> 16x32 tile per block)
  wmma_gemm_kernel<<<N_NODES/16, 64, 0, stream>>>(o1, w2t, h2, 128, 32);
  alpha_kernel<<<blk(N_NODES,T),T,0,stream>>>(h2, attS2, attD2, as2, ad2, 1, 32);
  edge_max_kernel<<<blk(E_TOT,T),T,0,stream>>>(ei, as2, ad2, e2, m2, 1);
  decode_max_kernel<<<blk(N_NODES,T),T,0,stream>>>(m2, N_NODES);
  edge_sum_kernel<<<blk(E_TOT,T),T,0,stream>>>(ei, (const float*)m2, e2, s2, 1);
  edge_agg_kernel<<<blk(E_TOT,T),T,0,stream>>>(ei, e2, s2, h2, o2, 1, 32);

  // ---------------- log_softmax -> d_out ------------------------------
  logsoftmax_kernel<<<blk(N_NODES,T),T,0,stream>>>(o2, b2, out);
}
